// CorrEncoderLayer_4827543241306
// MI455X (gfx1250) — matrix operand order repaired and checked by reading, not compiled, not run on hardware
//
#include <hip/hip_runtime.h>
#include <cstdint>

// ---------------------------------------------------------------------------
// CDNA5 / gfx1250 Autoformer encoder layer (CorrEncoderLayer)
// bf16 WMMA GEMMs (async LDS pipeline) + packed-real FFT correlation +
// top-k delay aggregation + moving-average decomposition.
// ---------------------------------------------------------------------------

typedef __bf16 bf16_t;
typedef __attribute__((ext_vector_type(16))) __bf16 v16bf;
typedef __attribute__((ext_vector_type(8)))  float  v8f;

struct alignas(16) U128 { unsigned int x, y, z, w; };
struct alignas(32) U256 { U128 lo, hi; };

#define PI_F 3.14159265358979f

constexpr int B_    = 8;
constexpr int L_    = 2048;
constexpr int D_    = 512;
constexpr int DFF_  = 2048;
constexpr int MR_   = B_ * L_;     // 16384 rows
constexpr int TOPK_ = 15;          // int(2*ln(2048)) = 15

// ---- gfx1250 async global->LDS path (guarded; falls back to reg staging) ---
#if defined(__HIP_DEVICE_COMPILE__) && __has_builtin(__builtin_amdgcn_global_load_async_to_lds_b128)
#define USE_ASYNC_LDS 1
#else
#define USE_ASYNC_LDS 0
#endif

#if USE_ASYNC_LDS
#if __has_builtin(__builtin_amdgcn_s_wait_asynccnt)
#define WAIT_ASYNC(n) __builtin_amdgcn_s_wait_asynccnt(n)
#else
#define WAIT_ASYNC(n) asm volatile("s_wait_asynccnt %0" :: "i"(n) : "memory")
#endif
// builtin signature (from clang diagnostic): param0 = int4 addrspace(1)*,
// param1 = int4 addrspace(3)*, then two integer-constant args (offset, cpol).
typedef int v4i_vs __attribute__((vector_size(16)));
#define AS1_PTR(p) ((__attribute__((address_space(1))) v4i_vs*)(uintptr_t)(p))
#define AS3_PTR(p) ((__attribute__((address_space(3))) v4i_vs*)(uintptr_t)(p))
#endif

// ---------------------------------------------------------------------------
// fp32 -> bf16 cast (grid-stride)
// ---------------------------------------------------------------------------
__global__ void cast_f32_bf16(const float* __restrict__ in, bf16_t* __restrict__ out, int n)
{
    int i = blockIdx.x * blockDim.x + threadIdx.x;
    const int stride = gridDim.x * blockDim.x;
    for (; i < n; i += stride) out[i] = (bf16_t)in[i];
}

// ---------------------------------------------------------------------------
// fp32 (K,N) -> bf16 (N,K) transposed cast for weights (tiled via LDS).
// Removes all transpose scatter from the GEMM hot loop.
// ---------------------------------------------------------------------------
__global__ __launch_bounds__(256)
void cast_transpose_bf16(const float* __restrict__ in, bf16_t* __restrict__ out,
                         int K, int N)
{
    __shared__ float tile[32][33];
    const int k0 = blockIdx.y * 32;
    const int n0 = blockIdx.x * 32;
    const int tx = threadIdx.x & 31;
    const int ty = threadIdx.x >> 5;       // 0..7
    #pragma unroll
    for (int r = ty; r < 32; r += 8)
        tile[r][tx] = in[(size_t)(k0 + r) * N + (n0 + tx)];
    __syncthreads();
    #pragma unroll
    for (int r = ty; r < 32; r += 8)
        out[(size_t)(n0 + r) * K + (k0 + tx)] = (bf16_t)tile[tx][r];
}

// ---------------------------------------------------------------------------
// bf16 WMMA GEMM:  C(M,N) = A(M,K) @ B(K,N), B given pre-transposed (N,K).
//   MODE 0: out fp32, transposed store q/k -> [(b*512+n)*2048 + l], + bias[n]
//   MODE 1: out fp32 row-major, + bias[n]                    (v)
//   MODE 2: out fp32 row-major, + bias[n] + res[m][n]        (Wo + residual)
//   MODE 3: out bf16 row-major, relu                         (FFN layer 1)
//   MODE 4: out fp32 row-major, + res[m][n]                  (FFN layer 2 + s1)
// Block tile 128x128, BK=32. 8 waves (4 M x 2 N), wave tile 32x64 =
// 2x4 v_wmma_f32_16x16x32_bf16 per K step. Double-buffered LDS with
// async global->LDS copies (ASYNCcnt) when available.
// ---------------------------------------------------------------------------
constexpr int BM = 128, BN = 128, BK = 32, LPAD = 8;

template<int MODE>
__global__ __launch_bounds__(256)
void gemm_bf16_wmma(const bf16_t* __restrict__ A, const bf16_t* __restrict__ BT,
                    int N, int K,
                    const float* __restrict__ bias, const float* __restrict__ res,
                    float* __restrict__ outF, bf16_t* __restrict__ outB)
{
    __shared__ bf16_t sA[2][BM][BK + LPAD];
    __shared__ bf16_t sB[2][BN][BK + LPAD];

    const int tid   = threadIdx.x;
    const int lane  = tid & 31;
    const int wave  = tid >> 5;            // 0..7
    const int wm    = wave & 3;            // 4 waves along M (32 rows each)
    const int wn    = wave >> 2;           // 2 waves along N (64 cols each)
    const int bm0   = blockIdx.y * BM;
    const int bn0   = blockIdx.x * BN;
    const int mlo   = lane & 15;
    const int khalf = lane >> 4;

    const int srow  = tid >> 2;            // 0..63 (4 threads per row)
    const int scol  = (tid & 3) << 3;      // 0,8,16,24

    v8f acc[2][4];
    #pragma unroll
    for (int i = 0; i < 2; ++i)
        #pragma unroll
        for (int j = 0; j < 4; ++j)
            acc[i][j] = (v8f){0.f,0.f,0.f,0.f,0.f,0.f,0.f,0.f};

    const bf16_t* aBase = A  + (size_t)(bm0 + srow) * K + scol;
    const bf16_t* bBase = BT + (size_t)(bn0 + srow) * K + scol;
    const size_t  aStep = (size_t)64 * K;
    const int nk = K / BK;

    // fragment compute on one LDS buffer
    auto compute_tile = [&](int buf) {
        v16bf af[2], bf4[4];
        #pragma unroll
        for (int i = 0; i < 2; ++i) {
            const int r = wm * 32 + i * 16 + mlo;
            U256 t;
            t.lo = *(const U128*)&sA[buf][r][khalf * 8];
            t.hi = *(const U128*)&sA[buf][r][khalf * 8 + 16];
            af[i] = __builtin_bit_cast(v16bf, t);
        }
        #pragma unroll
        for (int j = 0; j < 4; ++j) {
            const int n = wn * 64 + j * 16 + mlo;
            U256 t;
            t.lo = *(const U128*)&sB[buf][n][khalf * 8];
            t.hi = *(const U128*)&sB[buf][n][khalf * 8 + 16];
            bf4[j] = __builtin_bit_cast(v16bf, t);
        }
        #pragma unroll
        for (int i = 0; i < 2; ++i)
            #pragma unroll
            for (int j = 0; j < 4; ++j)
                acc[i][j] = __builtin_amdgcn_wmma_f32_16x16x32_bf16(
                    false, af[i], false, bf4[j], (short)0, acc[i][j], false, false);
    };

#if USE_ASYNC_LDS
    auto issue_tile = [&](int it) {
        const int buf = it & 1;
        const bf16_t* ga = aBase + (size_t)it * BK;
        const bf16_t* gb = bBase + (size_t)it * BK;
        __builtin_amdgcn_global_load_async_to_lds_b128(AS1_PTR(ga),
            AS3_PTR(&sA[buf][srow][scol]), 0, 0);
        __builtin_amdgcn_global_load_async_to_lds_b128(AS1_PTR(ga + aStep),
            AS3_PTR(&sA[buf][srow + 64][scol]), 0, 0);
        __builtin_amdgcn_global_load_async_to_lds_b128(AS1_PTR(gb),
            AS3_PTR(&sB[buf][srow][scol]), 0, 0);
        __builtin_amdgcn_global_load_async_to_lds_b128(AS1_PTR(gb + aStep),
            AS3_PTR(&sB[buf][srow + 64][scol]), 0, 0);
    };
    issue_tile(0);
    for (int it = 0; it < nk; ++it) {
        if (it + 1 < nk) { issue_tile(it + 1); WAIT_ASYNC(4); }
        else             { WAIT_ASYNC(0); }
        __syncthreads();                 // tile `it` resident for all waves
        compute_tile(it & 1);
        __syncthreads();                 // reads done before buf is re-filled
    }
#else
    // register-staged double buffer fallback
    U128 ra0, ra1, rb0, rb1;
    {
        const bf16_t* ga = aBase;
        const bf16_t* gb = bBase;
        ra0 = *(const U128*)ga;  ra1 = *(const U128*)(ga + aStep);
        rb0 = *(const U128*)gb;  rb1 = *(const U128*)(gb + aStep);
        *(U128*)&sA[0][srow][scol]      = ra0;
        *(U128*)&sA[0][srow + 64][scol] = ra1;
        *(U128*)&sB[0][srow][scol]      = rb0;
        *(U128*)&sB[0][srow + 64][scol] = rb1;
    }
    __syncthreads();
    for (int it = 0; it < nk; ++it) {
        if (it + 1 < nk) {
            const bf16_t* ga = aBase + (size_t)(it + 1) * BK;
            const bf16_t* gb = bBase + (size_t)(it + 1) * BK;
            ra0 = *(const U128*)ga;  ra1 = *(const U128*)(ga + aStep);
            rb0 = *(const U128*)gb;  rb1 = *(const U128*)(gb + aStep);
            if (it + 2 < nk) {
                __builtin_prefetch(ga + BK, 0, 3);
                __builtin_prefetch(gb + BK, 0, 3);
            }
        }
        compute_tile(it & 1);
        __syncthreads();
        if (it + 1 < nk) {
            const int nbuf = (it + 1) & 1;
            *(U128*)&sA[nbuf][srow][scol]      = ra0;
            *(U128*)&sA[nbuf][srow + 64][scol] = ra1;
            *(U128*)&sB[nbuf][srow][scol]      = rb0;
            *(U128*)&sB[nbuf][srow + 64][scol] = rb1;
        }
        __syncthreads();
    }
#endif

    // ---- epilogue: C/D layout: VGPR r -> M = khalf*8 + r, N = lane&15 ----
    #pragma unroll
    for (int i = 0; i < 2; ++i) {
        #pragma unroll
        for (int j = 0; j < 4; ++j) {
            #pragma unroll
            for (int r = 0; r < 8; ++r) {
                const int gm = bm0 + wm * 32 + i * 16 + khalf * 8 + r;
                const int gn = bn0 + wn * 64 + j * 16 + mlo;
                const float v = acc[i][j][r];
                if (MODE == 0) {
                    const int b = gm >> 11, l = gm & (L_ - 1);
                    outF[((size_t)(b * D_ + gn)) * L_ + l] = v + bias[gn];
                } else if (MODE == 1) {
                    outF[(size_t)gm * N + gn] = v + bias[gn];
                } else if (MODE == 2) {
                    outF[(size_t)gm * N + gn] = v + bias[gn] + res[(size_t)gm * N + gn];
                } else if (MODE == 3) {
                    outB[(size_t)gm * N + gn] = (bf16_t)fmaxf(v, 0.0f);
                } else {
                    outF[(size_t)gm * N + gn] = v + res[(size_t)gm * N + gn];
                }
            }
        }
    }
}

// ---------------------------------------------------------------------------
// 2048-pt radix-2 Stockham (self-sorting) FFT in LDS, 128 threads.
// sign = -1 forward, +1 inverse. Returns final buffer index (0 or 1).
// ---------------------------------------------------------------------------
__device__ __forceinline__
int fft2048(float* r0, float* i0, float* r1, float* i1, int tid, float sign)
{
    float* br[2] = { r0, r1 };
    float* bi[2] = { i0, i1 };
    int cur = 0, n = 2048, sstr = 1;
    #pragma unroll 1
    for (int st = 0; st < 11; ++st) {
        const int m = n >> 1;
        const float* sr = br[cur]; const float* si = bi[cur];
        float* dr = br[cur ^ 1];   float* di = bi[cur ^ 1];
        for (int f = tid; f < 1024; f += 128) {
            const int pp = f >> st;            // f / sstr
            const int qq = f & (sstr - 1);
            float sv, cv;
            __sincosf(sign * 2.0f * PI_F * (float)pp / (float)n, &sv, &cv);
            const float ar = sr[qq + sstr * pp],        ai = si[qq + sstr * pp];
            const float xr0 = sr[qq + sstr * (pp + m)], xi0 = si[qq + sstr * (pp + m)];
            dr[qq + sstr * 2 * pp] = ar + xr0;
            di[qq + sstr * 2 * pp] = ai + xi0;
            const float xr = ar - xr0, xi = ai - xi0;
            dr[qq + sstr * (2 * pp + 1)] = xr * cv - xi * sv;
            di[qq + sstr * (2 * pp + 1)] = xi * cv + xr * sv;
        }
        __syncthreads();
        cur ^= 1; n >>= 1; sstr <<= 1;
    }
    return cur;
}

// ---------------------------------------------------------------------------
// Packed-real forward FFT per (b,row) pair:  z = q_row + i*k_row.
// Extract Qf,Kf by Hermitian split, write spectral product
//   Re = qr*kr, Im = -(qi*ki)  in-place over the consumed q/k rows.
// ---------------------------------------------------------------------------
__global__ __launch_bounds__(128)
void fft_corr_spec(float* __restrict__ q, float* __restrict__ k)
{
    __shared__ float r0[2048], i0[2048], r1[2048], i1[2048];
    const int p   = blockIdx.x;            // b*512 + row
    const int tid = threadIdx.x;
    float* __restrict__ qrow = q + (size_t)p * L_;
    float* __restrict__ krow = k + (size_t)p * L_;

    for (int i = tid; i < L_; i += 128) { r0[i] = qrow[i]; i0[i] = krow[i]; }
    __syncthreads();

    const int cur = fft2048(r0, i0, r1, i1, tid, -1.0f);
    const float* zr = cur ? r1 : r0;
    const float* zi = cur ? i1 : i0;

    for (int j = tid; j <= L_ / 2; j += 128) {
        const int jn = (L_ - j) & (L_ - 1);
        const float a = zr[j], b = zi[j], an = zr[jn], bn = zi[jn];
        const float qr = 0.5f * (a + an);
        const float qi = 0.5f * (b - bn);
        const float kr = 0.5f * (b + bn);
        const float ki = -0.5f * (a - an);
        qrow[j] = qr * kr;        // Re(spec) partial for this (h,d) row
        krow[j] = -(qi * ki);     // Im(spec) partial
    }
}

// ---------------------------------------------------------------------------
// Deterministic tree reduction over the 512 (h,d) rows per batch, per bin.
// ---------------------------------------------------------------------------
__global__ __launch_bounds__(128)
void reduce_spec(const float* __restrict__ q, const float* __restrict__ k,
                 float* __restrict__ ssum)
{
    __shared__ float red[128];
    const int b = blockIdx.x / 1025;
    const int j = blockIdx.x % 1025;
    const int tid = threadIdx.x;
    float sr = 0.f, si = 0.f;
    for (int r = tid; r < D_; r += 128) {
        const size_t off = ((size_t)(b * D_ + r)) * L_ + j;
        sr += q[off];
        si += k[off];
    }
    red[tid] = sr; __syncthreads();
    for (int s = 64; s > 0; s >>= 1) { if (tid < s) red[tid] += red[tid + s]; __syncthreads(); }
    if (tid == 0) ssum[(size_t)(b * 1025 + j) * 2] = red[0];
    __syncthreads();
    red[tid] = si; __syncthreads();
    for (int s = 64; s > 0; s >>= 1) { if (tid < s) red[tid] += red[tid + s]; __syncthreads(); }
    if (tid == 0) ssum[(size_t)(b * 1025 + j) * 2 + 1] = red[0];
}

// ---------------------------------------------------------------------------
// Hermitian-extend the 1025 accumulated bins, inverse FFT, scale by
// 1/(N * 512)  (irfft norm * mean over h*d), write mean_value(b, l).
// ---------------------------------------------------------------------------
__global__ __launch_bounds__(128)
void ifft_mean(const float* __restrict__ ssum, float* __restrict__ meanc)
{
    __shared__ float r0[2048], i0[2048], r1[2048], i1[2048];
    const int b = blockIdx.x, tid = threadIdx.x;
    for (int j = tid; j < 2048; j += 128) {
        const int js = (j <= 1024) ? j : 2048 - j;
        float re = ssum[(size_t)(b * 1025 + js) * 2];
        float im = ssum[(size_t)(b * 1025 + js) * 2 + 1];
        if (j > 1024) im = -im;
        r0[j] = re; i0[j] = im;
    }
    __syncthreads();
    const int cur = fft2048(r0, i0, r1, i1, tid, +1.0f);
    const float* rr = cur ? r1 : r0;
    const float sc = 1.0f / (2048.0f * 512.0f);
    for (int l = tid; l < 2048; l += 128) meanc[b * 2048 + l] = rr[l] * sc;
}

// ---------------------------------------------------------------------------
// Top-15 delays per batch + softmax of their correlation values.
// ---------------------------------------------------------------------------
__global__ __launch_bounds__(256)
void topk_softmax(const float* __restrict__ meanc, int* __restrict__ dly,
                  float* __restrict__ wts)
{
    __shared__ float vals[2048];
    __shared__ float rv[256];
    __shared__ int   ri[256];
    __shared__ float topv[16];
    const int b = blockIdx.x, tid = threadIdx.x;
    for (int i = tid; i < 2048; i += 256) vals[i] = meanc[b * 2048 + i];
    __syncthreads();
    for (int t = 0; t < TOPK_; ++t) {
        float best = -__builtin_inff(); int bi = 0;
        for (int i = tid; i < 2048; i += 256) {
            const float v = vals[i];
            if (v > best) { best = v; bi = i; }
        }
        rv[tid] = best; ri[tid] = bi; __syncthreads();
        for (int s = 128; s > 0; s >>= 1) {
            if (tid < s) {
                if (rv[tid + s] > rv[tid] ||
                    (rv[tid + s] == rv[tid] && ri[tid + s] < ri[tid])) {
                    rv[tid] = rv[tid + s]; ri[tid] = ri[tid + s];
                }
            }
            __syncthreads();
        }
        if (tid == 0) { topv[t] = rv[0]; dly[b * 16 + t] = ri[0]; vals[ri[0]] = -__builtin_inff(); }
        __syncthreads();
    }
    if (tid == 0) {
        float mx = -__builtin_inff();
        for (int t = 0; t < TOPK_; ++t) mx = fmaxf(mx, topv[t]);
        float s = 0.f, e[TOPK_];
        for (int t = 0; t < TOPK_; ++t) { e[t] = __expf(topv[t] - mx); s += e[t]; }
        const float inv = 1.0f / s;
        for (int t = 0; t < TOPK_; ++t) wts[b * 16 + t] = e[t] * inv;
    }
}

// ---------------------------------------------------------------------------
// agg(b,l,:) = sum_t w[t] * v(b, (l+delay[t]) mod L, :), stored bf16 for Wo GEMM
// ---------------------------------------------------------------------------
__global__ __launch_bounds__(256)
void roll_aggregate(const float* __restrict__ v, const int* __restrict__ dly,
                    const float* __restrict__ wts, bf16_t* __restrict__ aggb)
{
    __shared__ float w[TOPK_];
    __shared__ int   dl[TOPK_];
    const size_t idx = (size_t)blockIdx.x * 256 + threadIdx.x;
    const int b = (int)(idx >> 20);                 // L_*D_ = 2^20 per batch
    if (threadIdx.x < TOPK_) {
        w[threadIdx.x]  = wts[b * 16 + threadIdx.x];
        dl[threadIdx.x] = dly[b * 16 + threadIdx.x];
    }
    __syncthreads();
    const int rem = (int)(idx & ((1u << 20) - 1));
    const int l = rem >> 9;
    const int d = rem & (D_ - 1);
    float acc = 0.f;
    #pragma unroll
    for (int t = 0; t < TOPK_; ++t) {
        int lt = l + dl[t]; if (lt >= L_) lt -= L_;
        acc += w[t] * v[((size_t)(b * L_ + lt)) * D_ + d];
    }
    aggb[idx] = (bf16_t)acc;
}

// ---------------------------------------------------------------------------
// Moving-average decomposition: s = x - trend, trend = mean of centered
// 25-wide window with clamped (edge-replicated) indices. Optional bf16 copy.
// ---------------------------------------------------------------------------
__global__ __launch_bounds__(256)
void ma_decomp_dual(const float* __restrict__ x, float* __restrict__ s,
                    bf16_t* __restrict__ sb)
{
    const size_t idx = (size_t)blockIdx.x * 256 + threadIdx.x;
    const int b   = (int)(idx >> 20);
    const int rem = (int)(idx & ((1u << 20) - 1));
    const int l = rem >> 9;
    const int d = rem & (D_ - 1);
    float sum = 0.f;
    #pragma unroll
    for (int o = -12; o <= 12; ++o) {
        int m = l + o;
        m = m < 0 ? 0 : (m > L_ - 1 ? L_ - 1 : m);
        sum += x[((size_t)(b * L_ + m)) * D_ + d];
    }
    const float val = x[idx] - sum * (1.0f / 25.0f);
    s[idx] = val;
    if (sb) sb[idx] = (bf16_t)val;
}

// ---------------------------------------------------------------------------
extern "C" void kernel_launch(void* const* d_in, const int* in_sizes, int n_in,
                              void* d_out, int out_size, void* d_ws, size_t ws_size,
                              hipStream_t stream)
{
    (void)in_sizes; (void)n_in; (void)out_size; (void)ws_size;
    const float* x  = (const float*)d_in[0];
    const float* Wq = (const float*)d_in[1];
    const float* bq = (const float*)d_in[2];
    const float* Wk = (const float*)d_in[3];
    const float* bk = (const float*)d_in[4];
    const float* Wv = (const float*)d_in[5];
    const float* bv = (const float*)d_in[6];
    const float* Wo = (const float*)d_in[7];
    const float* bo = (const float*)d_in[8];
    const float* W1 = (const float*)d_in[9];
    const float* W2 = (const float*)d_in[10];
    float* out = (float*)d_out;

    char* ws = (char*)d_ws;
    size_t off = 0;
    auto alloc = [&](size_t bytes) -> void* {
        void* p = ws + off;
        off += (bytes + 255) & ~(size_t)255;
        return p;
    };

    bf16_t* WBq = (bf16_t*)alloc((size_t)D_ * D_ * 2);     // transposed (N,K)
    bf16_t* WBk = (bf16_t*)alloc((size_t)D_ * D_ * 2);
    bf16_t* WBv = (bf16_t*)alloc((size_t)D_ * D_ * 2);
    bf16_t* WBo = (bf16_t*)alloc((size_t)D_ * D_ * 2);
    bf16_t* WB1 = (bf16_t*)alloc((size_t)D_ * DFF_ * 2);   // (2048,512)
    bf16_t* WB2 = (bf16_t*)alloc((size_t)DFF_ * D_ * 2);   // (512,2048)
    bf16_t* XB  = (bf16_t*)alloc((size_t)MR_ * D_ * 2);    // reused: AGGB, S1B
    float*  Q   = (float*)alloc((size_t)MR_ * D_ * 4);     // reused: XR
    float*  Kb  = (float*)alloc((size_t)MR_ * D_ * 4);     // reused: S1
    float*  V   = (float*)alloc((size_t)MR_ * D_ * 4);     // reused: Z
    float*  SSUM  = (float*)alloc((size_t)B_ * 1025 * 2 * 4);
    float*  MEANC = (float*)alloc((size_t)B_ * 2048 * 4);
    int*    DLY   = (int*)alloc((size_t)B_ * 16 * 4);
    float*  WTS   = (float*)alloc((size_t)B_ * 16 * 4);
    bf16_t* H     = (bf16_t*)alloc((size_t)MR_ * DFF_ * 2);

    bf16_t* AGGB = XB;     // valid after QKV GEMMs consumed XB
    bf16_t* S1B  = XB;     // valid after Wo GEMM consumed AGGB
    float*  XR   = Q;      // valid after spectra reduced out of Q
    float*  S1   = Kb;     // valid after spectra reduced out of Kb
    float*  Z    = V;      // valid after roll_aggregate consumed V

    // 1) weight casts with transpose, activation cast
    dim3 tD(D_ / 32, D_ / 32);          // (16,16)
    cast_transpose_bf16<<<tD, 256, 0, stream>>>(Wq, WBq, D_, D_);
    cast_transpose_bf16<<<tD, 256, 0, stream>>>(Wk, WBk, D_, D_);
    cast_transpose_bf16<<<tD, 256, 0, stream>>>(Wv, WBv, D_, D_);
    cast_transpose_bf16<<<tD, 256, 0, stream>>>(Wo, WBo, D_, D_);
    dim3 t1(DFF_ / 32, D_ / 32);        // W1 (512,2048) -> (2048,512)
    cast_transpose_bf16<<<t1, 256, 0, stream>>>(W1, WB1, D_, DFF_);
    dim3 t2(D_ / 32, DFF_ / 32);        // W2 (2048,512) -> (512,2048)
    cast_transpose_bf16<<<t2, 256, 0, stream>>>(W2, WB2, DFF_, D_);
    cast_f32_bf16<<<4096, 256, 0, stream>>>(x, XB, MR_ * D_);

    // 2) Q/K/V projections (q,k transposed to (b, h*dk, L) for FFT rows)
    dim3 gD(D_ / BN, MR_ / BM);                       // (4, 128)
    gemm_bf16_wmma<0><<<gD, 256, 0, stream>>>(XB, WBq, D_, D_, bq, nullptr, Q,  nullptr);
    gemm_bf16_wmma<0><<<gD, 256, 0, stream>>>(XB, WBk, D_, D_, bk, nullptr, Kb, nullptr);
    gemm_bf16_wmma<1><<<gD, 256, 0, stream>>>(XB, WBv, D_, D_, bv, nullptr, V,  nullptr);

    // 3) spectral correlation: forward FFTs -> per-row products -> reduce
    fft_corr_spec<<<B_ * D_, 128, 0, stream>>>(Q, Kb);
    reduce_spec<<<B_ * 1025, 128, 0, stream>>>(Q, Kb, SSUM);
    ifft_mean<<<B_, 128, 0, stream>>>(SSUM, MEANC);
    topk_softmax<<<B_, 256, 0, stream>>>(MEANC, DLY, WTS);

    // 4) delay aggregation + output projection + residual
    roll_aggregate<<<(MR_ * D_) / 256, 256, 0, stream>>>(V, DLY, WTS, AGGB);
    gemm_bf16_wmma<2><<<gD, 256, 0, stream>>>(AGGB, WBo, D_, D_, bo, x, XR, nullptr);

    // 5) decomp 1 (fp32 s1 + bf16 copy for FFN GEMM)
    ma_decomp_dual<<<(MR_ * D_) / 256, 256, 0, stream>>>(XR, S1, S1B);

    // 6) FFN: relu(s1 @ W1) @ W2 + s1
    dim3 gF(DFF_ / BN, MR_ / BM);                     // (16, 128)
    gemm_bf16_wmma<3><<<gF, 256, 0, stream>>>(S1B, WB1, DFF_, D_, nullptr, nullptr, nullptr, H);
    gemm_bf16_wmma<4><<<gD, 256, 0, stream>>>(H, WB2, D_, DFF_, nullptr, S1, Z, nullptr);

    // 7) decomp 2 -> output
    ma_decomp_dual<<<(MR_ * D_) / 256, 256, 0, stream>>>(Z, out, nullptr);
}
